// Head_28484223107446
// MI455X (gfx1250) — matrix operand order repaired and checked by reading, not compiled, not run on hardware
//
#include <hip/hip_runtime.h>
#include <hip/hip_bf16.h>

#define B_ 8
#define T_ 4096
#define C_ 1024
#define H_ 64

typedef __attribute__((ext_vector_type(16))) __bf16 v16bf;
typedef __attribute__((ext_vector_type(8)))  float  v8f;

#define WMMA_BF16(A, Bm, Cm) \
  __builtin_amdgcn_wmma_f32_16x16x32_bf16(false, (A), false, (Bm), (short)0, (Cm), false, false)

// Load a 16-element bf16 fragment as two 16B chunks (each 16B-aligned).
__device__ __forceinline__ v16bf load2x16(const __bf16* p0, const __bf16* p1) {
  union { uint4 u[2]; v16bf v; } t;
  t.u[0] = *(const uint4*)p0;
  t.u[1] = *(const uint4*)p1;
  return t.v;
}

// XOR-butterfly over 16 lanes via ds_swizzle (group-of-32 mode: and=0x1f, xor=XM).
template <int XM>
__device__ __forceinline__ float swz_xor(float v) {
  return __int_as_float(
      __builtin_amdgcn_ds_swizzle(__float_as_int(v), (0x1f | (XM << 10))));
}
__device__ __forceinline__ float red16_max(float v) {
  v = fmaxf(v, swz_xor<1>(v));
  v = fmaxf(v, swz_xor<2>(v));
  v = fmaxf(v, swz_xor<4>(v));
  v = fmaxf(v, swz_xor<8>(v));
  return v;
}
__device__ __forceinline__ float red16_sum(float v) {
  v += swz_xor<1>(v);
  v += swz_xor<2>(v);
  v += swz_xor<4>(v);
  v += swz_xor<8>(v);
  return v;
}

// ---------------------------------------------------------------------------
// Kernel 1: convert + transpose weights.  Wt[which][h][c] = W_which[c][h] (bf16)
// ---------------------------------------------------------------------------
__global__ void convert_weights(const float* __restrict__ Wq,
                                const float* __restrict__ Wk,
                                const float* __restrict__ Wv,
                                __bf16* __restrict__ Wt) {
  int idx = blockIdx.x * blockDim.x + threadIdx.x;
  if (idx >= 3 * H_ * C_) return;
  int which = idx >> 16;     // / (64*1024)
  int rem   = idx & 65535;
  int h     = rem >> 10;     // / 1024
  int c     = rem & 1023;
  const float* src = (which == 0) ? Wq : (which == 1) ? Wk : Wv;
  Wt[idx] = (__bf16)src[(size_t)c * H_ + h];
}

// ---------------------------------------------------------------------------
// Kernel 2: fused QKV projection.  One wave -> 16 rows x 64 cols of q,k,v.
// q,k stored [B*T, H] bf16 (q pre-scaled by C^-0.5); v stored transposed
// [B][H][T] bf16 so P*V B-fragments are contiguous.
// ---------------------------------------------------------------------------
__global__ __launch_bounds__(256) void qkv_gemm(const float* __restrict__ x,
                                                const __bf16* __restrict__ Wt,
                                                __bf16* __restrict__ qb,
                                                __bf16* __restrict__ kb,
                                                __bf16* __restrict__ vT) {
  const int lane = threadIdx.x & 31;
  const int wave = threadIdx.x >> 5;
  const int l16  = lane & 15;
  const int hi   = (lane >> 4) & 1;
  const int t0   = (blockIdx.x * 8 + wave) * 16;   // global row tile (over B*T)

  v8f accq[4] = {};
  v8f acck[4] = {};
  v8f accv[4] = {};

  const float* xrow = x + (size_t)(t0 + l16) * C_;

  for (int cb = 0; cb < C_; cb += 32) {
    // A fragment (16x32, bf16): row = l16, k contiguous at cb+16*hi .. cb+16*hi+15
    const float* p0 = xrow + cb + 16 * hi;
    const float* p1 = xrow + cb + 16 * hi + 8;
    float4 f0 = ((const float4*)p0)[0];
    float4 f1 = ((const float4*)p0)[1];
    float4 f2 = ((const float4*)p1)[0];
    float4 f3 = ((const float4*)p1)[1];
    v16bf a;
    a[0]  = (__bf16)f0.x; a[1]  = (__bf16)f0.y; a[2]  = (__bf16)f0.z; a[3]  = (__bf16)f0.w;
    a[4]  = (__bf16)f1.x; a[5]  = (__bf16)f1.y; a[6]  = (__bf16)f1.z; a[7]  = (__bf16)f1.w;
    a[8]  = (__bf16)f2.x; a[9]  = (__bf16)f2.y; a[10] = (__bf16)f2.z; a[11] = (__bf16)f2.w;
    a[12] = (__bf16)f3.x; a[13] = (__bf16)f3.y; a[14] = (__bf16)f3.z; a[15] = (__bf16)f3.w;

#pragma unroll
    for (int ns = 0; ns < 4; ++ns) {
      const size_t wo = (size_t)(ns * 16 + l16) * C_ + cb + 16 * hi;
      const __bf16* wq = Wt + 0 * 65536 + wo;
      const __bf16* wk = Wt + 1 * 65536 + wo;
      const __bf16* wv = Wt + 2 * 65536 + wo;
      v16bf bq = load2x16(wq, wq + 8);
      accq[ns] = WMMA_BF16(a, bq, accq[ns]);
      v16bf bk = load2x16(wk, wk + 8);
      acck[ns] = WMMA_BF16(a, bk, acck[ns]);
      v16bf bv = load2x16(wv, wv + 8);
      accv[ns] = WMMA_BF16(a, bv, accv[ns]);
    }
  }

  const int bIdx = t0 / T_;
  const int tt0  = t0 - bIdx * T_;          // 16-row tiles never cross batches
#pragma unroll
  for (int ns = 0; ns < 4; ++ns) {
    const int h = ns * 16 + l16;
#pragma unroll
    for (int r = 0; r < 8; ++r) {
      const size_t t = (size_t)(t0 + r + 8 * hi);
      qb[t * H_ + h] = (__bf16)(accq[ns][r] * 0.03125f);  // fold C^-0.5
      kb[t * H_ + h] = (__bf16)(acck[ns][r]);
    }
    union { __bf16 e[8]; uint4 u; } pk;
#pragma unroll
    for (int r = 0; r < 8; ++r) pk.e[r] = (__bf16)(accv[ns][r]);
    *(uint4*)(vT + (size_t)bIdx * H_ * T_ + (size_t)h * T_ + tt0 + 8 * hi) = pk.u;
  }
}

// ---------------------------------------------------------------------------
// Kernel 3: causal flash attention.  Block = 8 waves; wave -> 16-query tile.
// Streams 32-key blocks; S via 4 WMMAs, online softmax, P*V via 4 WMMAs.
// ---------------------------------------------------------------------------
__global__ __launch_bounds__(256) void flash_attn(const __bf16* __restrict__ qb,
                                                  const __bf16* __restrict__ kbm,
                                                  const __bf16* __restrict__ vT,
                                                  float* __restrict__ out) {
  __shared__ __align__(16) __bf16 Plds[8][16][32];   // per-wave P staging (1KB each)

  const int lane = threadIdx.x & 31;
  const int wave = threadIdx.x >> 5;
  const int l16  = lane & 15;
  const int hi   = (lane >> 4) & 1;
  const int bIdx = blockIdx.x >> 5;      // / 32
  const int qblk = blockIdx.x & 31;
  const int qt   = qblk * 128 + wave * 16;

  const __bf16* qbase = qb  + (size_t)bIdx * T_ * H_;
  const __bf16* kbase = kbm + (size_t)bIdx * T_ * H_;
  const __bf16* vbase = vT  + (size_t)bIdx * H_ * T_;
  float*        obase = out + (size_t)bIdx * T_ * H_;

  // Q A-fragments: 16 queries x 64 h -> two 16x32 fragments
  const __bf16* qrow = qbase + (size_t)(qt + l16) * H_;
  v16bf aq0 = load2x16(qrow + 16 * hi,      qrow + 16 * hi + 8);   // h 0..31
  v16bf aq1 = load2x16(qrow + 32 + 16 * hi, qrow + 40 + 16 * hi);   // h 32..63

  v8f o0 = {}, o1 = {}, o2 = {}, o3 = {};
  float m_r[8], l_r[8];
#pragma unroll
  for (int r = 0; r < 8; ++r) { m_r[r] = -1e30f; l_r[r] = 0.0f; }

  const int kbmax = (qt + 15) >> 5;
  for (int kbi = 0; kbi <= kbmax; ++kbi) {
    const int key0 = kbi * 32 + l16;
    const int key1 = key0 + 16;

    // K^T B-fragments: per lane col = key, k = h contiguous
    const __bf16* kp0 = kbase + (size_t)key0 * H_ + 16 * hi;
    const __bf16* kp1 = kbase + (size_t)key1 * H_ + 16 * hi;
    v16bf bk00 = load2x16(kp0,      kp0 + 8);   // keys 0..15, h 0..31
    v16bf bk01 = load2x16(kp0 + 32, kp0 + 40);  // keys 0..15, h 32..63
    v16bf bk10 = load2x16(kp1,      kp1 + 8);
    v16bf bk11 = load2x16(kp1 + 32, kp1 + 40);

    v8f s0 = {}, s1 = {};
    s0 = WMMA_BF16(aq0, bk00, s0);
    s0 = WMMA_BF16(aq1, bk01, s0);
    s1 = WMMA_BF16(aq0, bk10, s1);
    s1 = WMMA_BF16(aq1, bk11, s1);

    if (kbi < kbmax) {  // prefetch next key block (global_prefetch_b8)
      __builtin_prefetch(kbase + (size_t)(key0 + 32) * H_, 0, 1);
      __builtin_prefetch(vbase + (size_t)l16 * T_ + (size_t)(kbi + 1) * 32, 0, 1);
    }

    // Online softmax per query row (row r -> query qt + r + 8*hi)
    float corr_r[8];
#pragma unroll
    for (int r = 0; r < 8; ++r) {
      const int qi = qt + r + 8 * hi;
      float s0v = (key0 <= qi) ? s0[r] : -1e30f;
      float s1v = (key1 <= qi) ? s1[r] : -1e30f;
      float rm   = red16_max(fmaxf(s0v, s1v));
      float mnew = fmaxf(m_r[r], rm);
      float corr = __expf(m_r[r] - mnew);
      float p0   = __expf(s0v - mnew);
      float p1   = __expf(s1v - mnew);
      float rs   = red16_sum(p0 + p1);
      l_r[r]    = l_r[r] * corr + rs;
      m_r[r]    = mnew;
      corr_r[r] = corr;
      Plds[wave][r + 8 * hi][l16]      = (__bf16)p0;   // D-layout -> LDS
      Plds[wave][r + 8 * hi][l16 + 16] = (__bf16)p1;
    }
#pragma unroll
    for (int r = 0; r < 8; ++r) {
      o0[r] *= corr_r[r]; o1[r] *= corr_r[r];
      o2[r] *= corr_r[r]; o3[r] *= corr_r[r];
    }

    // Wave-private LDS RAW fence (no cross-wave barrier: trip counts differ)
    asm volatile("s_wait_dscnt 0" ::: "memory");

    // P as A-fragment (16 queries x 32 keys)
    const __bf16* pr = &Plds[wave][l16][0];
    v16bf ap = load2x16(pr + 16 * hi, pr + 16 * hi + 8);

    // V B-fragments from transposed V: per lane col = h, k = key contiguous
    const __bf16* vp = vbase + (size_t)l16 * T_ + (size_t)kbi * 32 + 16 * hi;
    v16bf bv0 = load2x16(vp,                    vp + 8);
    v16bf bv1 = load2x16(vp + (size_t)16 * T_,  vp + (size_t)16 * T_ + 8);
    v16bf bv2 = load2x16(vp + (size_t)32 * T_,  vp + (size_t)32 * T_ + 8);
    v16bf bv3 = load2x16(vp + (size_t)48 * T_,  vp + (size_t)48 * T_ + 8);
    o0 = WMMA_BF16(ap, bv0, o0);
    o1 = WMMA_BF16(ap, bv1, o1);
    o2 = WMMA_BF16(ap, bv2, o2);
    o3 = WMMA_BF16(ap, bv3, o3);
  }

#pragma unroll
  for (int r = 0; r < 8; ++r) {
    const float inv = 1.0f / l_r[r];
    const size_t ro = (size_t)(qt + r + 8 * hi) * H_;
    obase[ro + l16]      = o0[r] * inv;
    obase[ro + 16 + l16] = o1[r] * inv;
    obase[ro + 32 + l16] = o2[r] * inv;
    obase[ro + 48 + l16] = o3[r] * inv;
  }
}

// ---------------------------------------------------------------------------
extern "C" void kernel_launch(void* const* d_in, const int* in_sizes, int n_in,
                              void* d_out, int out_size, void* d_ws, size_t ws_size,
                              hipStream_t stream) {
  (void)in_sizes; (void)n_in; (void)out_size; (void)ws_size;
  const float* x  = (const float*)d_in[0];
  const float* Wq = (const float*)d_in[1];
  const float* Wk = (const float*)d_in[2];
  const float* Wv = (const float*)d_in[3];
  float* out = (float*)d_out;

  char* ws = (char*)d_ws;
  __bf16* Wt = (__bf16*)(ws);                      // 3*64*1024*2   =   393,216 B
  __bf16* qb = (__bf16*)(ws + 393216);             // B*T*H*2       = 4,194,304 B
  __bf16* kb = (__bf16*)(ws + 4587520);            // B*T*H*2
  __bf16* vT = (__bf16*)(ws + 8781824);            // B*H*T*2  (transposed V)

  convert_weights<<<768, 256, 0, stream>>>(Wq, Wk, Wv, Wt);
  qkv_gemm<<<256, 256, 0, stream>>>(x, Wt, qb, kb, vT);          // 2048 waves
  flash_attn<<<B_ * (T_ / 128), 256, 0, stream>>>(qb, kb, vT, out);
}
